// MGDSGU_12524124635329
// MI455X (gfx1250) — compile-verified, matrix operand order checked by reading, not run on hardware
//
#include <hip/hip_runtime.h>
#include <hip/hip_bf16.h>

// ---------------------------------------------------------------------------
// MGDSGU forward for MI455X (gfx1250): bf16 WMMA GEMMs + atomic scatter-mean.
// Dead code eliminated: sage_input and the gamma branch are unused in the
// reference; h_N(post) = h + tanh(Wb1 h + b1 + Wb2 hN0 + b2).
// GEMM: 64x64 tile / 4 waves, double-buffered LDS, software-pipelined so
// global b128 loads overlap the v_wmma_f32_16x16x32_bf16 stream.
// ---------------------------------------------------------------------------

typedef __attribute__((ext_vector_type(16))) __bf16 v16bf;
typedef __attribute__((ext_vector_type(8)))  float  v8f;

#define D        512
#define DWORDS   (D / 2)      // dwords per bf16 feature row
#define ROWQ     (DWORDS / 4) // uint4 per bf16 feature row (64)
#define TILE     64
#define NTHREADS 128          // 4 wave32
#define NSTEPS   32           // (K/32 chunks) x (2 operand pairs)

union Frag16 { v16bf v; unsigned int u[8]; };

__device__ __forceinline__ unsigned short f2bf(float f) {
  unsigned int u = __float_as_uint(f);
  u += 0x7FFFu + ((u >> 16) & 1u);          // round-to-nearest-even
  return (unsigned short)(u >> 16);
}

// ---------------------------------------------------------------------------
// C = act( [Cin] + A1 @ W1^T + A2 @ W2^T + bias1 + bias2 )
// A: Mpad x 512 bf16 row-major (rows >= M zero-padded), W: 512x512 bf16 (NxK).
// act: 0 = none, 1 = sigmoid, 2 = tanh.
// ---------------------------------------------------------------------------
__global__ __launch_bounds__(NTHREADS) void
gemm_pair(const unsigned short* __restrict__ A1, const unsigned short* __restrict__ W1,
          const unsigned short* __restrict__ A2, const unsigned short* __restrict__ W2,
          const float* __restrict__ bias1, const float* __restrict__ bias2,
          const float* __restrict__ Cin, float* __restrict__ Cout,
          int M, int act)
{
  __shared__ uint4 As4[2][TILE * 4];       // 64 rows x 32 bf16 (4 uint4 / row)
  __shared__ uint4 Bs4[2][TILE * 4];

  const int tid  = threadIdx.x;
  const int wave = tid >> 5;
  const int lane = tid & 31;
  const int half = lane >> 4;
  const int l16  = lane & 15;
  const int mblk = blockIdx.x * TILE;
  const int nblk = blockIdx.y * TILE;

  const uint4* Au[2] = { (const uint4*)A1, (const uint4*)A2 };
  const uint4* Wu[2] = { (const uint4*)W1, (const uint4*)W2 };

  // per-thread staging coordinates: thread moves uint4 #tid and #tid+128
  const int r0 = tid >> 2;                 // rows 0..31
  const int c4 = tid & 3;
  const int r1 = r0 + 32;                  // rows 32..63

  uint4 ra0, ra1, rb0, rb1;

#define GLOAD(s)                                                        \
  { int p_ = (s) & 1; int kq_ = ((s) >> 1) * 4;                         \
    ra0 = Au[p_][(size_t)(mblk + r0) * ROWQ + kq_ + c4];                \
    rb0 = Wu[p_][(size_t)(nblk + r0) * ROWQ + kq_ + c4];                \
    ra1 = Au[p_][(size_t)(mblk + r1) * ROWQ + kq_ + c4];                \
    rb1 = Wu[p_][(size_t)(nblk + r1) * ROWQ + kq_ + c4]; }

#define LSTORE(buf)                                                     \
  { As4[buf][tid] = ra0;       Bs4[buf][tid] = rb0;                     \
    As4[buf][tid + 128] = ra1; Bs4[buf][tid + 128] = rb1; }

  v8f acc[4];
#pragma unroll
  for (int t = 0; t < 4; ++t)
#pragma unroll
    for (int v = 0; v < 8; ++v) acc[t][v] = 0.0f;

  GLOAD(0);
  LSTORE(0);

#pragma unroll 2
  for (int s = 0; s < NSTEPS; ++s) {
    __syncthreads();                       // buf[s&1] ready; buf[!(s&1)] free
    const int cur = s & 1;

    if (s + 1 < NSTEPS) GLOAD(s + 1);      // in flight during the WMMAs below
    if (s + 2 < NSTEPS) {                  // prefetch A stream 2 steps ahead
      int p_ = (s + 2) & 1; int kq_ = ((s + 2) >> 1) * 4;
      __builtin_prefetch(&Au[p_][(size_t)(mblk + r0) * ROWQ + kq_ + c4], 0, 1);
    }

    const unsigned int* As = (const unsigned int*)As4[cur];
    const unsigned int* Bs = (const unsigned int*)Bs4[cur];

    // A fragment: 16-bit A 16x32 layout (ISA 7.12.2):
    //   VGPR v<4: K = 2v + 8*half ; v>=4: K = 16 + 2(v-4) + 8*half
    Frag16 a;
#pragma unroll
    for (int v = 0; v < 8; ++v) {
      int dw = ((v < 4) ? v : v + 4) + 4 * half;   // dword index within row
      a.u[v] = As[(16 * wave + l16) * 16 + dw];
    }
#pragma unroll
    for (int t = 0; t < 4; ++t) {
      Frag16 b;
#pragma unroll
      for (int v = 0; v < 8; ++v) {
        int dw = ((v < 4) ? v : v + 4) + 4 * half;
        b.u[v] = Bs[(16 * t + l16) * 16 + dw];
      }
      acc[t] = __builtin_amdgcn_wmma_f32_16x16x32_bf16(
          false, a.v, false, b.v, (short)0, acc[t], false, false);
    }

    if (s + 1 < NSTEPS) LSTORE(!cur);      // s_wait_loadcnt lands after WMMAs
  }
#undef GLOAD
#undef LSTORE

  // C/D layout: VGPR v, lanes 0-15 -> M=v, lanes 16-31 -> M=v+8 ; N = lane%16
  const int mloc = 16 * wave + 8 * half;
#pragma unroll
  for (int t = 0; t < 4; ++t) {
    int n = nblk + 16 * t + l16;
    float bsum = (bias1 ? bias1[n] : 0.0f) + (bias2 ? bias2[n] : 0.0f);
#pragma unroll
    for (int v = 0; v < 8; ++v) {
      int gm = mblk + mloc + v;
      if (gm < M) {
        long idx = (long)gm * D + n;
        float val = acc[t][v] + bsum;
        if (Cin) val += Cin[idx];
        if (act == 1)      val = 1.0f / (1.0f + __expf(-val));
        else if (act == 2) val = tanhf(val);
        Cout[idx] = val;
      }
    }
  }
}

// ------------------------------ edge / elementwise -------------------------
__global__ void zero_f32(float* __restrict__ p, int n) {
  int i = blockIdx.x * 256 + threadIdx.x;
  if (i < n) p[i] = 0.0f;
}

__global__ void zero_bf16(unsigned short* __restrict__ p, int n) {
  int i = blockIdx.x * 256 + threadIdx.x;
  if (i < n) p[i] = 0;
}

__global__ void cvt_bf16(const float* __restrict__ in, unsigned short* __restrict__ out, int n) {
  int i = blockIdx.x * 256 + threadIdx.x;
  if (i < n) out[i] = f2bf(in[i]);
}

__global__ void deg_count(const long long* __restrict__ ei, float* __restrict__ deg, int E) {
  int e = blockIdx.x * 256 + threadIdx.x;
  if (e < E) atomicAdd(&deg[(int)ei[(long)E + e]], 1.0f);
}

__global__ void inv_deg_k(const float* __restrict__ deg, float* __restrict__ inv, int n) {
  int i = blockIdx.x * 256 + threadIdx.x;
  if (i < n) { float d = deg[i]; inv[i] = (d > 0.0f) ? (1.0f / d) : 1.0f; }
}

// One block per edge, 256 threads x 2 floats: acc[dst] += feat[src]
__global__ void scatter_add(const float* __restrict__ feat, const long long* __restrict__ ei,
                            float* __restrict__ acc, int E) {
  int e = blockIdx.x;
  int s = (int)ei[e];
  int d = (int)ei[(long)E + e];
  const float* fs = feat + (long)s * D;
  float*       ad = acc  + (long)d * D;
  int j = threadIdx.x;
  atomicAdd(&ad[j],        fs[j]);
  atomicAdd(&ad[j + 256],  fs[j + 256]);
}

__global__ void scale_cvt(const float* __restrict__ acc, const float* __restrict__ inv,
                          unsigned short* __restrict__ out, int n) {
  int i = blockIdx.x * 256 + threadIdx.x;
  if (i < n) out[i] = f2bf(acc[i] * inv[i >> 9]);
}

__global__ void add_k(const float* __restrict__ a, const float* __restrict__ b,
                      float* __restrict__ o, int n) {
  int i = blockIdx.x * 256 + threadIdx.x;
  if (i < n) o[i] = a[i] + b[i];
}

__global__ void mul_inplace(float* __restrict__ a, const float* __restrict__ b, int n) {
  int i = blockIdx.x * 256 + threadIdx.x;
  if (i < n) a[i] *= b[i];
}

__global__ void gru_out(const float* __restrict__ z, const float* __restrict__ hn,
                        const float* __restrict__ ht, float* __restrict__ o, int n) {
  int i = blockIdx.x * 256 + threadIdx.x;
  if (i < n) { float zz = z[i]; o[i] = (1.0f - zz) * hn[i] + zz * ht[i]; }
}

// ---------------------------------------------------------------------------
extern "C" void kernel_launch(void* const* d_in, const int* in_sizes, int n_in,
                              void* d_out, int out_size, void* d_ws, size_t ws_size,
                              hipStream_t stream)
{
  (void)n_in; (void)out_size; (void)ws_size;

  const float*     x  = (const float*)d_in[0];
  const float*     h  = (const float*)d_in[1];
  const long long* ei = (const long long*)d_in[2];
  const int M    = in_sizes[0] / D;
  const int E    = in_sizes[2] / 2;
  const int NF   = M * D;
  const int Mpad = (M + TILE - 1) & ~(TILE - 1);
  const int NFp  = Mpad * D;
  const int NPAD = NFp - NF;                  // zero-padded tail of bf16 bufs

  // params flattened in dict insertion order (each sage: Wl, bl, Wr)
  const float* Wl_hid = (const float*)d_in[6];
  const float* bl_hid = (const float*)d_in[7];
  const float* Wr_hid = (const float*)d_in[8];
  const float* Wl_sx  = (const float*)d_in[9];
  const float* bl_sx  = (const float*)d_in[10];
  const float* Wr_sx  = (const float*)d_in[11];
  const float* Wl_sh  = (const float*)d_in[12];
  const float* bl_sh  = (const float*)d_in[13];
  const float* Wr_sh  = (const float*)d_in[14];
  const float* Wl_ux  = (const float*)d_in[15];
  const float* bl_ux  = (const float*)d_in[16];
  const float* Wr_ux  = (const float*)d_in[17];
  const float* Wl_uh  = (const float*)d_in[18];
  const float* bl_uh  = (const float*)d_in[19];
  const float* Wr_uh  = (const float*)d_in[20];
  const float* Wl_cx  = (const float*)d_in[21];
  const float* bl_cx  = (const float*)d_in[22];
  const float* Wr_cx  = (const float*)d_in[23];
  const float* Wl_ch  = (const float*)d_in[24];
  const float* bl_ch  = (const float*)d_in[25];
  const float* Wr_ch  = (const float*)d_in[26];
  // d_in[27..30] = W_gamma1/2 (dead code), d_in[3..5] = sage_input (dead code)
  const float* Wb1 = (const float*)d_in[31];
  const float* bb1 = (const float*)d_in[32];
  const float* Wb2 = (const float*)d_in[33];
  const float* bb2 = (const float*)d_in[34];

  // ---- carve workspace -----------------------------------------------------
  char* wptr = (char*)d_ws;
  auto carve = [&](size_t bytes) -> void* {
    void* p = (void*)wptr;
    wptr += (bytes + 255) & ~(size_t)255;
    return p;
  };
  float* F0 = (float*)carve((size_t)NF * 4);   // scatter accumulator (reused 4x)
  float* F1 = (float*)carve((size_t)NF * 4);   // hN0 -> h_N
  float* F2 = (float*)carve((size_t)NF * 4);   // beta -> h_tilde
  float* F3 = (float*)carve((size_t)NF * 4);   // r -> rh
  float* F4 = (float*)carve((size_t)NF * 4);   // z
  unsigned short* B0 = (unsigned short*)carve((size_t)NFp * 2); // x_bf
  unsigned short* B1 = (unsigned short*)carve((size_t)NFp * 2); // x_agg_bf
  unsigned short* B2 = (unsigned short*)carve((size_t)NFp * 2); // h_bf -> rh_bf
  unsigned short* B3 = (unsigned short*)carve((size_t)NFp * 2); // agg scratch
  unsigned short* B4 = (unsigned short*)carve((size_t)NFp * 2); // hN0_bf -> hN_bf
  float* deg = (float*)carve((size_t)M * 4);
  float* inv = (float*)carve((size_t)M * 4);
  unsigned short* WB[16];
  for (int i = 0; i < 16; ++i) WB[i] = (unsigned short*)carve((size_t)D * D * 2);

  const float* wsrc[16] = { Wl_hid, Wr_hid, Wl_sx, Wr_sx, Wl_sh, Wr_sh,
                            Wl_ux, Wr_ux, Wl_uh, Wr_uh, Wl_cx, Wr_cx,
                            Wl_ch, Wr_ch, Wb1, Wb2 };

  const int nfB = (NF + 255) / 256;
  const int WN  = D * D;
  const dim3 gg(Mpad / TILE, D / TILE);

  // ---- zero the 64-row padding tails of all bf16 activation buffers -------
  if (NPAD > 0) {
    int pb = (NPAD + 255) / 256;
    zero_bf16<<<pb, 256, 0, stream>>>(B0 + NF, NPAD);
    zero_bf16<<<pb, 256, 0, stream>>>(B1 + NF, NPAD);
    zero_bf16<<<pb, 256, 0, stream>>>(B2 + NF, NPAD);
    zero_bf16<<<pb, 256, 0, stream>>>(B3 + NF, NPAD);
    zero_bf16<<<pb, 256, 0, stream>>>(B4 + NF, NPAD);
  }

  // ---- weight + input conversion to bf16 ----------------------------------
  for (int i = 0; i < 16; ++i)
    cvt_bf16<<<(WN + 255) / 256, 256, 0, stream>>>(wsrc[i], WB[i], WN);
  cvt_bf16<<<nfB, 256, 0, stream>>>(x, B0, NF);
  cvt_bf16<<<nfB, 256, 0, stream>>>(h, B2, NF);

  // ---- degrees -------------------------------------------------------------
  zero_f32<<<(M + 255) / 256, 256, 0, stream>>>(deg, M);
  deg_count<<<(E + 255) / 256, 256, 0, stream>>>(ei, deg, E);
  inv_deg_k<<<(M + 255) / 256, 256, 0, stream>>>(deg, inv, M);

  // ---- x_agg -> B1, h_agg -> B3 -------------------------------------------
  zero_f32<<<nfB, 256, 0, stream>>>(F0, NF);
  scatter_add<<<E, 256, 0, stream>>>(x, ei, F0, E);
  scale_cvt<<<nfB, 256, 0, stream>>>(F0, inv, B1, NF);

  zero_f32<<<nfB, 256, 0, stream>>>(F0, NF);
  scatter_add<<<E, 256, 0, stream>>>(h, ei, F0, E);
  scale_cvt<<<nfB, 256, 0, stream>>>(F0, inv, B3, NF);

  // ---- hN0 = sage_hidden(h, h_agg) -> F1 ----------------------------------
  gemm_pair<<<gg, NTHREADS, 0, stream>>>(B3, WB[0], B2, WB[1], bl_hid, nullptr,
                                         nullptr, F1, M, 0);
  cvt_bf16<<<nfB, 256, 0, stream>>>(F1, B4, NF);

  // ---- beta = tanh(Wb1 h + bb1 + Wb2 hN0 + bb2) -> F2 ---------------------
  gemm_pair<<<gg, NTHREADS, 0, stream>>>(B2, WB[14], B4, WB[15], bb1, bb2,
                                         nullptr, F2, M, 2);

  // ---- h_N = h + beta -> F1, bf16 -> B4 -----------------------------------
  add_k<<<nfB, 256, 0, stream>>>(h, F2, F1, NF);
  cvt_bf16<<<nfB, 256, 0, stream>>>(F1, B4, NF);

  // ---- hN_agg -> B3 -------------------------------------------------------
  zero_f32<<<nfB, 256, 0, stream>>>(F0, NF);
  scatter_add<<<E, 256, 0, stream>>>(F1, ei, F0, E);
  scale_cvt<<<nfB, 256, 0, stream>>>(F0, inv, B3, NF);

  // ---- r = sigmoid(sage_single_x + sage_single_h) -> F3 -------------------
  gemm_pair<<<gg, NTHREADS, 0, stream>>>(B1, WB[2], B0, WB[3], bl_sx, nullptr,
                                         nullptr, F3, M, 0);
  gemm_pair<<<gg, NTHREADS, 0, stream>>>(B3, WB[4], B4, WB[5], bl_sh, nullptr,
                                         F3, F3, M, 1);

  // ---- z = sigmoid(sage_update_x + sage_update_h) -> F4 -------------------
  gemm_pair<<<gg, NTHREADS, 0, stream>>>(B1, WB[6], B0, WB[7], bl_ux, nullptr,
                                         nullptr, F4, M, 0);
  gemm_pair<<<gg, NTHREADS, 0, stream>>>(B3, WB[8], B4, WB[9], bl_uh, nullptr,
                                         F4, F4, M, 1);

  // ---- rh = r * h_N (in place F3), bf16 -> B2, rh_agg -> B3 ---------------
  mul_inplace<<<nfB, 256, 0, stream>>>(F3, F1, NF);
  cvt_bf16<<<nfB, 256, 0, stream>>>(F3, B2, NF);
  zero_f32<<<nfB, 256, 0, stream>>>(F0, NF);
  scatter_add<<<E, 256, 0, stream>>>(F3, ei, F0, E);
  scale_cvt<<<nfB, 256, 0, stream>>>(F0, inv, B3, NF);

  // ---- h_tilde = tanh(sage_candidate_x + sage_candidate_h) -> F2 ----------
  gemm_pair<<<gg, NTHREADS, 0, stream>>>(B1, WB[10], B0, WB[11], bl_cx, nullptr,
                                         nullptr, F2, M, 0);
  gemm_pair<<<gg, NTHREADS, 0, stream>>>(B3, WB[12], B2, WB[13], bl_ch, nullptr,
                                         F2, F2, M, 2);

  // ---- out = (1-z)*h_N + z*h_tilde ----------------------------------------
  gru_out<<<nfB, 256, 0, stream>>>(F4, F1, F2, (float*)d_out, NF);
}
